// Sequence_2465311228251
// MI455X (gfx1250) — matrix operand order repaired and checked by reading, not compiled, not run on hardware
//
#include <hip/hip_runtime.h>

typedef _Float16 v16h __attribute__((ext_vector_type(16)));
typedef _Float16 h8   __attribute__((ext_vector_type(8)));
typedef float    v8f  __attribute__((ext_vector_type(8)));

#define T_STEPS 512
#define HID     128
#define PITCH   136                 // halves per row in LDS h buffers (17*8)
#define HBUFSZ  (16 * PITCH)        // one h buffer (halves)

__device__ __forceinline__ float fast_sig(float x) {
    return __builtin_amdgcn_rcpf(1.0f + __expf(-x));
}
__device__ __forceinline__ float fast_tanh(float x) {
    return 2.0f * __builtin_amdgcn_rcpf(1.0f + __expf(-2.0f * x)) - 1.0f;
}

// Load a 16x32 f16 A-fragment (M=row lanes, K contiguous per lane half: kb*32 + hi*16 + [0..15]) from an LDS h buffer.
__device__ __forceinline__ v16h loadA(const _Float16* hb, int kb, int m, int hi) {
    const h8* p = reinterpret_cast<const h8*>(hb + m * PITCH + kb * 32 + hi * 16);
    union { v16h v; h8 h[2]; } u;
    u.h[0] = p[0];   // K = kb*32 + hi*16 + [0..7]
    u.h[1] = p[1];   // K = kb*32 + hi*16 + 8 + [0..7]
    return u.v;
}

// Load a pre-swizzled 32-byte B-fragment lane chunk from LDS.
__device__ __forceinline__ v16h loadB(const _Float16* base) {
    return *reinterpret_cast<const v16h*>(base);
}

// Load 16 contiguous f32 weights (one W row slice) and convert to a f16 B-fragment lane chunk.
__device__ __forceinline__ v16h loadWfrag(const float* __restrict__ W, int n, int k0) {
    const float* src = W + n * HID + k0;
    v16h r;
#pragma unroll
    for (int i = 0; i < 16; ++i) r[i] = (_Float16)src[i];
    return r;
}

__global__ __launch_bounds__(256) void lstm2_persistent_kernel(
    const float* __restrict__ x,
    const float* __restrict__ Wih1, const float* __restrict__ Whh1,
    const float* __restrict__ bih1, const float* __restrict__ bhh1,
    const float* __restrict__ Wih2, const float* __restrict__ Whh2,
    const float* __restrict__ bih2, const float* __restrict__ bhh2,
    const float* __restrict__ Wl,   const float* __restrict__ bl,
    float* __restrict__ out)
{
    // 256 KB: Wih2 + Whh2 as f16 B-fragments, pre-swizzled.
    // Fragment (mtx, j, kb) lane l chunk (16 halves) at halves offset
    //   ((mtx*128 + j*4 + kb)*32 + l) * 16
    __shared__ __align__(16) _Float16 wlds[2 * 128 * 512];
    __shared__ __align__(16) float    xtile[16 * T_STEPS];
    __shared__ __align__(16) _Float16 hbuf1[2 * HBUFSZ];   // double-buffered h1
    __shared__ __align__(16) _Float16 hbuf2[2 * HBUFSZ];   // double-buffered h2

    const int tid    = threadIdx.x;
    const int wave   = tid >> 5;
    const int lane   = tid & 31;
    const int l15    = lane & 15;
    const int hi     = lane >> 4;        // which half of the wave (K/M split)
    const int hi8    = hi * 8;
    const int batch0 = blockIdx.x * 16;
    const int col    = wave * 16 + l15;  // hidden unit owned by this (wave, lane)
    const int laneoff = lane * 16;       // halves offset of this lane's chunk within a fragment

    // ---- Layer-1 recurrent weights resident in VGPRs (4 gates x 4 K-blocks = 128 VGPRs) ----
    v16h W1f[4][4];
#pragma unroll
    for (int g = 0; g < 4; ++g) {
        const int n = g * 128 + col;          // row of [4H, H] weight
#pragma unroll
        for (int kb = 0; kb < 4; ++kb) {
            const int k0 = kb * 32 + hi * 16; // lane half covers K 0-15 or 16-31 of block
            W1f[g][kb] = loadWfrag(Whh1, n, k0);
        }
    }

    // Per-(gate, hidden) scalars: fused biases and the IN=1 input weight column.
    float b1s[4], b2s[4], w1s[4];
#pragma unroll
    for (int g = 0; g < 4; ++g) {
        const int idx = g * 128 + col;
        b1s[g] = bih1[idx] + bhh1[idx];
        b2s[g] = bih2[idx] + bhh2[idx];
        w1s[g] = Wih1[idx];
    }

    // ---- Stage layer-2 weights into LDS as pre-swizzled f16 fragments ----
    // chunk id c = mtx*4096 + j*128 + kb*32 + l  (8192 chunks of 16 halves)
    for (int c = tid; c < 2 * 128 * 32; c += 256) {
        const int mtx = c >> 12;
        const int cc  = c & 4095;
        const int j   = cc >> 7;
        const int kb  = (cc >> 5) & 3;
        const int l   = cc & 31;
        const float* W = mtx ? Whh2 : Wih2;
        const int n  = j * 16 + (l & 15);
        const int k0 = kb * 32 + (l >> 4) * 16;
        const float* src = W + n * HID + k0;
        _Float16* dst = &wlds[(size_t)c * 16];
#pragma unroll
        for (int i = 0; i < 16; ++i) dst[i] = (_Float16)src[i];
    }

    // ---- Cooperative staging: x tile to LDS, zero h buffers ----
    for (int i = tid; i < 16 * T_STEPS; i += 256) {
        const int r = i >> 9;             // row 0..15
        const int t = i & (T_STEPS - 1);
        xtile[r * T_STEPS + t] = x[(size_t)(batch0 + r) * T_STEPS + t];
    }
    for (int i = tid; i < 2 * HBUFSZ; i += 256) {
        hbuf1[i] = (_Float16)0.0f;
        hbuf2[i] = (_Float16)0.0f;
    }

    float c1[8], c2[8];
#pragma unroll
    for (int v = 0; v < 8; ++v) { c1[v] = 0.0f; c2[v] = 0.0f; }

    const v8f zc = {0.0f, 0.0f, 0.0f, 0.0f, 0.0f, 0.0f, 0.0f, 0.0f};

    __syncthreads();

    for (int t = 0; t < T_STEPS; ++t) {
        const int p = t & 1;
        const _Float16* h1r = hbuf1 + p * HBUFSZ;
        _Float16*       h1w = hbuf1 + (1 - p) * HBUFSZ;
        const _Float16* h2r = hbuf2 + p * HBUFSZ;
        _Float16*       h2w = hbuf2 + (1 - p) * HBUFSZ;

        v8f acc1[4], acc2[4];
        float hn[8];

        // ===== Hoisted: acc2 = h2_old @ Whh2^T (independent of layer 1; C=0 on first WMMA) =====
        {
            const v16h a = loadA(h2r, 0, l15, hi);
#pragma unroll
            for (int g = 0; g < 4; ++g) {
                const v16h b = loadB(&wlds[(128 * 32 + ((g * 8 + wave) * 4 + 0) * 32) * 16 + laneoff]);
                acc2[g] = __builtin_amdgcn_wmma_f32_16x16x32_f16(
                    false, a, false, b, (short)0, zc, false, false);
            }
        }
#pragma unroll
        for (int kb = 1; kb < 4; ++kb) {
            const v16h a = loadA(h2r, kb, l15, hi);
#pragma unroll
            for (int g = 0; g < 4; ++g) {
                const v16h b = loadB(&wlds[(128 * 32 + ((g * 8 + wave) * 4 + kb) * 32) * 16 + laneoff]);
                acc2[g] = __builtin_amdgcn_wmma_f32_16x16x32_f16(
                    false, a, false, b, (short)0, acc2[g], false, false);
            }
        }

        // ===== Layer 1: gates = b1 + x_t*Wih1 + h1_old @ Whh1^T (weights in VGPRs) =====
#pragma unroll
        for (int g = 0; g < 4; ++g) {
#pragma unroll
            for (int v = 0; v < 8; ++v) {
                const float xm = xtile[(hi8 + v) * T_STEPS + t];  // LDS broadcast per lane-half
                acc1[g][v] = b1s[g] + xm * w1s[g];
            }
        }
#pragma unroll
        for (int kb = 0; kb < 4; ++kb) {
            const v16h a = loadA(h1r, kb, l15, hi);
#pragma unroll
            for (int g = 0; g < 4; ++g)
                acc1[g] = __builtin_amdgcn_wmma_f32_16x16x32_f16(
                    false, a, false, W1f[g][kb], (short)0, acc1[g], false, false);
        }
#pragma unroll
        for (int v = 0; v < 8; ++v) {
            const float ig = fast_sig(acc1[0][v]);
            const float fg = fast_sig(acc1[1][v]);
            const float gg = fast_tanh(acc1[2][v]);
            const float og = fast_sig(acc1[3][v]);
            c1[v] = fg * c1[v] + ig * gg;
            hn[v] = og * fast_tanh(c1[v]);
        }
#pragma unroll
        for (int v = 0; v < 8; ++v)
            h1w[(hi8 + v) * PITCH + col] = (_Float16)hn[v];
        __syncthreads();   // h1_new visible to all waves (old/new buffers disjoint)

        // ===== Layer 2 remainder: acc2 += h1_new @ Wih2^T (weights in LDS) =====
#pragma unroll
        for (int kb = 0; kb < 4; ++kb) {
            const v16h a = loadA(h1w, kb, l15, hi);
#pragma unroll
            for (int g = 0; g < 4; ++g) {
                const v16h b = loadB(&wlds[(((g * 8 + wave) * 4 + kb) * 32) * 16 + laneoff]);
                acc2[g] = __builtin_amdgcn_wmma_f32_16x16x32_f16(
                    false, a, false, b, (short)0, acc2[g], false, false);
            }
        }
#pragma unroll
        for (int v = 0; v < 8; ++v) {
            const float ig = fast_sig(acc2[0][v] + b2s[0]);
            const float fg = fast_sig(acc2[1][v] + b2s[1]);
            const float gg = fast_tanh(acc2[2][v] + b2s[2]);
            const float og = fast_sig(acc2[3][v] + b2s[3]);
            c2[v] = fg * c2[v] + ig * gg;
            hn[v] = og * fast_tanh(c2[v]);
        }
#pragma unroll
        for (int v = 0; v < 8; ++v)
            h2w[(hi8 + v) * PITCH + col] = (_Float16)hn[v];
        __syncthreads();   // h2_new visible; next step reads the other buffers
    }

    // ===== Epilogue: out[b] = h2[b,:] . Wl + bl =====
    // Last step (t = T-1, odd) wrote h2 into buffer 0.
    if (tid < 16) {
        const _Float16* h2f = hbuf2 + ((T_STEPS & 1) ? (1 - (T_STEPS - 1 & 1)) : (1 - ((T_STEPS - 1) & 1))) * HBUFSZ;
        // T_STEPS = 512 -> last write buffer index = 1 - (511 & 1) = 0
        h2f = hbuf2;
        float s = bl[0];
        for (int k = 0; k < HID; ++k)
            s += (float)h2f[tid * PITCH + k] * Wl[k];
        out[batch0 + tid] = s;
    }
}

extern "C" void kernel_launch(void* const* d_in, const int* in_sizes, int n_in,
                              void* d_out, int out_size, void* d_ws, size_t ws_size,
                              hipStream_t stream) {
    (void)in_sizes; (void)n_in; (void)d_ws; (void)ws_size; (void)out_size;
    const float* x    = (const float*)d_in[0];
    const float* Wih1 = (const float*)d_in[1];
    const float* Whh1 = (const float*)d_in[2];
    const float* bih1 = (const float*)d_in[3];
    const float* bhh1 = (const float*)d_in[4];
    const float* Wih2 = (const float*)d_in[5];
    const float* Whh2 = (const float*)d_in[6];
    const float* bih2 = (const float*)d_in[7];
    const float* bhh2 = (const float*)d_in[8];
    const float* Wl   = (const float*)d_in[9];
    const float* bl   = (const float*)d_in[10];
    float* out = (float*)d_out;

    dim3 grid(1024 / 16);   // 64 blocks, each owns a 16-row batch tile for all T
    dim3 block(256);        // 8 waves of 32
    lstm2_persistent_kernel<<<grid, block, 0, stream>>>(
        x, Wih1, Whh1, bih1, bhh1, Wih2, Whh2, bih2, bhh2, Wl, bl, out);
}
